// PSA_23167053594843
// MI455X (gfx1250) — compile-verified, matrix operand order checked by reading, not run on hardware
//
#include <hip/hip_runtime.h>

// ---------------------------------------------------------------------------
// Types for CDNA5 WMMA
// ---------------------------------------------------------------------------
typedef __attribute__((ext_vector_type(16))) __bf16  v16bf;
typedef __attribute__((ext_vector_type(8)))  float   v8f;
typedef __attribute__((ext_vector_type(2)))  float   v2f;
typedef __attribute__((ext_vector_type(4)))  int     i32x4;

union FragBF { v16bf v; uint4 q[2]; };

#define HW   4096   // 64*64
#define BATCH 8

// ---------------------------------------------------------------------------
// Async global->LDS (CDNA5 Tensor/Async path, ASYNCcnt). Guarded: falls back
// to synchronous VGPR staging if the builtin is absent on this toolchain.
// Prototype (from hipcc diagnostic): (int4 AS1*, int4 AS3*, imm offset, imm cpol)
// ---------------------------------------------------------------------------
#if defined(__gfx1250__) && __has_builtin(__builtin_amdgcn_global_load_async_to_lds_b128)
#define USE_ASYNC_LDS 1
#else
#define USE_ASYNC_LDS 0
#endif

#if USE_ASYNC_LDS
__device__ __forceinline__ void async_g2l_b128(const void* gsrc, const void* ldst) {
  // AS(1)/AS(3) pointers built via integer casts: AS3 ptr == low 32 bits of the
  // generic LDS address (ISA 10.2 aperture rule), AS1 ptr == generic address.
  auto gp = (__attribute__((address_space(1))) i32x4*)(unsigned long long)gsrc;
  auto lp = (__attribute__((address_space(3))) i32x4*)(unsigned)(unsigned long long)ldst;
  __builtin_amdgcn_global_load_async_to_lds_b128(gp, lp, 0, 0);
}
__device__ __forceinline__ void wait_async0() {
#if __has_builtin(__builtin_amdgcn_s_wait_asynccnt)
  __builtin_amdgcn_s_wait_asynccnt(0);
#else
  asm volatile("s_wait_asynccnt 0x0" ::: "memory");
#endif
}
#endif

__device__ __forceinline__ unsigned short f2bf(float v) {
  union { __bf16 h; unsigned short u; } c;
  c.h = (__bf16)v;
  return c.u;
}

// ---------------------------------------------------------------------------
// Prep kernels: weight f32 -> bf16 repack, BN(+bias)(+BN) fusion into scale/shift
// ---------------------------------------------------------------------------
__global__ __launch_bounds__(256) void f32_to_bf16_kernel(
    const float* __restrict__ in, unsigned short* __restrict__ out, int n)
{
  int i = blockIdx.x * 256 + threadIdx.x;
  if (i < n) out[i] = f2bf(in[i]);
}

// out: y = x*s + t  implementing  bn2(bn1(x + bias))   (bn2 optional, bias optional)
__global__ __launch_bounds__(256) void bn_fuse_kernel(
    const float* g1, const float* b1, const float* m1, const float* v1,
    const float* bias,
    const float* g2, const float* b2, const float* m2, const float* v2,
    float* s, float* t, int C)
{
  int c = blockIdx.x * 256 + threadIdx.x;
  if (c >= C) return;
  float inv = rsqrtf(v1[c] + 1e-5f);
  float sc  = g1[c] * inv;
  float bb  = bias ? bias[c] : 0.0f;
  float sh  = (bb - m1[c]) * sc + b1[c];
  if (g2) {
    float inv2 = rsqrtf(v2[c] + 1e-5f);
    float s2   = g2[c] * inv2;
    sh = (sh - m2[c]) * s2 + b2[c];
    sc = sc * s2;
  }
  s[c] = sc;
  t[c] = sh;
}

// ---------------------------------------------------------------------------
// Fused conv1x1 GEMM:  out[b,o,p] = epi( sum_c W[o,c] * pro(in[b,c_off+c,p]) )
//   pro: optional LIF spike (binary -> exact in bf16);  W pre-packed bf16
//   epi: per-channel scale/shift (fused BN(+bias)(+BN)), optional SiLU (rcp),
//        optional residual add (same indexing as out)
// Block: 256 thr = 8 waves; tile 128(O) x 128(P); wave does 32x64 (2x4 WMMA tiles)
// W tile staged via GLOBAL_LOAD_ASYNC_TO_LDS_B128 (pure copy, overlaps with the
// VALU spike/bf16 convert+transpose of the X tile).
// ---------------------------------------------------------------------------
__global__ __launch_bounds__(256) void gemm_bf16_kernel(
    const float* __restrict__ in, int Ctot, int c_off, int C,
    const unsigned short* __restrict__ W,          // [O][C] bf16 bits
    const float* __restrict__ scale, const float* __restrict__ shift,
    float* out, int Otot, int o_off,
    const float* residual,
    int spike, float thr, int silu)
{
  // row stride 40 halves (80 B): 16-byte alignment for b128/b32 LDS ops,
  // 20-bank step between rows to spread conflicts.
  __shared__ __align__(16) unsigned short Wl[128 * 40];
  __shared__ __align__(16) unsigned short Xl[128 * 40];

  const int t  = threadIdx.x;
  const int p0 = blockIdx.x * 128;
  const int o0 = blockIdx.y * 128;
  const int bz = blockIdx.z;

  const int wv = t >> 5, l = t & 31;
  const int ow = wv >> 1;          // O sub-strip (4 x 32 rows)
  const int pw = wv & 1;           // P sub-strip (2 x 64 cols)
  const int lm = l & 15;
  const int kb = (l < 16) ? 0 : 8; // K base for A/B fragments (bf16 16x32 layout)
  const int ro = (l < 16) ? 0 : 8; // row offset for C/D fragments

  // preload per-lane BN scale/shift (16 output channels per lane)
  float sc[2][8], sh[2][8];
  #pragma unroll
  for (int m = 0; m < 2; ++m)
    #pragma unroll
    for (int r = 0; r < 8; ++r) {
      const int o_rel = o0 + ow * 32 + m * 16 + ro + r;
      sc[m][r] = scale[o_rel];
      sh[m][r] = shift[o_rel];
    }

  v8f acc[2][4] = {};

  const int ksteps = C >> 5;
  for (int kt = 0; kt < ksteps; ++kt) {
    // ---- stage weights: 128 x 32 bf16 ----
#if USE_ASYNC_LDS
    #pragma unroll
    for (int it = 0; it < 2; ++it) {
      int f = t + it * 256;                 // 0..511 groups of 8 halves
      int r = f >> 2, g = f & 3;
      async_g2l_b128(&W[(size_t)(o0 + r) * C + kt * 32 + g * 8],
                     &Wl[r * 40 + g * 8]);
    }
#else
    #pragma unroll
    for (int it = 0; it < 2; ++it) {
      int f = t + it * 256;
      int r = f >> 2, g = f & 3;
      uint4 d = *(const uint4*)&W[(size_t)(o0 + r) * C + kt * 32 + g * 8];
      *(uint4*)&Wl[r * 40 + g * 8] = d;
    }
#endif
    // ---- stage activations transposed: Xl[p][k], spike+bf16 on the fly;
    //      two k-adjacent halves packed into one b32 LDS store ----
    #pragma unroll
    for (int it = 0; it < 2; ++it) {
      int task = t + it * 256;              // 0..511: 16 k-pairs x 32 p-groups
      int k  = (task >> 5) * 2;
      int p  = (task & 31) * 4;
      const size_t rb = ((size_t)bz * Ctot + c_off + kt * 32 + k) * HW + p0 + p;
      const float4 x0 = *(const float4*)&in[rb];
      const float4 x1 = *(const float4*)&in[rb + HW];
      float f0[4] = {x0.x, x0.y, x0.z, x0.w};
      float f1[4] = {x1.x, x1.y, x1.z, x1.w};
      #pragma unroll
      for (int j = 0; j < 4; ++j) {
        float a0 = f0[j], a1 = f1[j];
        if (spike) {                        // LIF, TAU=2
          a0 = (a0 * 0.5f - thr >= 0.0f) ? 1.0f : 0.0f;
          a1 = (a1 * 0.5f - thr >= 0.0f) ? 1.0f : 0.0f;
        }
        unsigned int packed = (unsigned int)f2bf(a0) | ((unsigned int)f2bf(a1) << 16);
        *(unsigned int*)&Xl[(p + j) * 40 + k] = packed;
      }
    }
#if USE_ASYNC_LDS
    wait_async0();
#endif
    __syncthreads();

    // ---- gather fragments (two aligned 16B runs each) and WMMA ----
    FragBF a[2], b[4];
    #pragma unroll
    for (int m = 0; m < 2; ++m) {
      const unsigned short* ap = &Wl[(ow * 32 + m * 16 + lm) * 40 + kb];
      a[m].q[0] = *(const uint4*)ap;          // K = kb .. kb+7
      a[m].q[1] = *(const uint4*)(ap + 16);   // K = kb+16 .. kb+23
    }
    #pragma unroll
    for (int n = 0; n < 4; ++n) {
      const unsigned short* bp = &Xl[(pw * 64 + n * 16 + lm) * 40 + kb];
      b[n].q[0] = *(const uint4*)bp;
      b[n].q[1] = *(const uint4*)(bp + 16);
    }
    #pragma unroll
    for (int m = 0; m < 2; ++m)
      #pragma unroll
      for (int n = 0; n < 4; ++n)
        acc[m][n] = __builtin_amdgcn_wmma_f32_16x16x32_bf16(
            false, a[m].v, false, b[n].v, (short)0, acc[m][n], false, false);

    __syncthreads();
  }

  // ---- epilogue: scale/shift (+SiLU via v_rcp) (+residual), f32 store ----
  #pragma unroll
  for (int m = 0; m < 2; ++m) {
    #pragma unroll
    for (int n = 0; n < 4; ++n) {
      const int p_g = p0 + pw * 64 + n * 16 + lm;
      #pragma unroll
      for (int r = 0; r < 8; ++r) {
        const int o_rel = o0 + ow * 32 + m * 16 + ro + r;
        float v = acc[m][n][r] * sc[m][r] + sh[m][r];
        if (silu) v = v * __builtin_amdgcn_rcpf(1.0f + __expf(-v));
        const size_t oi = ((size_t)bz * Otot + o_off + o_rel) * HW + p_g;
        if (residual) v += residual[oi];
        out[oi] = v;
      }
    }
  }
}

// ---------------------------------------------------------------------------
// Depthwise 3x3, SAME padding, NCHW 64x64 planes.
// ---------------------------------------------------------------------------
__global__ __launch_bounds__(256) void dwconv3x3_kernel(
    const float* __restrict__ in, const float* __restrict__ w,
    float* __restrict__ out, int C)
{
  size_t idx = (size_t)blockIdx.x * 256 + threadIdx.x;  // b*C*HW elements
  int    p   = (int)(idx & 4095);
  size_t bc  = idx >> 12;
  int    c   = (int)(bc % C);
  int    h   = p >> 6, x = p & 63;
  const float* wc    = w + c * 9;
  const float* plane = in + (bc << 12);
  float s = 0.0f;
  #pragma unroll
  for (int dy = -1; dy <= 1; ++dy) {
    int hh = h + dy;
    if (hh < 0 || hh > 63) continue;
    #pragma unroll
    for (int dx = -1; dx <= 1; ++dx) {
      int xx = x + dx;
      if (xx < 0 || xx > 63) continue;
      s += wc[(dy + 1) * 3 + (dx + 1)] * plane[hh * 64 + xx];
    }
  }
  out[idx] = s;
}

// ---------------------------------------------------------------------------
// Spiking spatial attention, one wave per (b,c) 64x64 plane.
//   KV[w,u] = sum_h K[h,w] * spike(V[h,u]/2 - 1)     (f32 WMMA 16x16x4)
//   A [h,u] = sum_w Q[h,w] * KV[w,u];  out = spike(A*0.125/2 - 0.5)  (binary)
// K/Q planes copied via async global->LDS; spiked V converted through VGPRs.
// ---------------------------------------------------------------------------
__global__ __launch_bounds__(32) void attn_kernel(
    const float* __restrict__ Q, const float* __restrict__ K,
    const float* __restrict__ V, float* __restrict__ out)
{
  __shared__ float Al[64 * 64];   // K plane, later reused for Q plane
  __shared__ float Vl[64 * 64];   // spiked V
  __shared__ float KVl[64 * 64];

  const int    l    = threadIdx.x;
  const size_t base = (size_t)blockIdx.x << 12;   // (b*256+c)*4096
  const int    lm   = l & 15;
  const int    kb2  = (l < 16) ? 0 : 2;           // f32 A/B frag K base
  const int    ro   = (l < 16) ? 0 : 8;

  // K plane: pure copy (async if available), V: spike through VGPRs
#if USE_ASYNC_LDS
  for (int i = 0; i < 32; ++i) {
    int f = l + i * 32;
    async_g2l_b128((const float4*)(K + base) + f, (float4*)Al + f);
  }
#else
  for (int i = 0; i < 32; ++i) {
    int f = l + i * 32;
    ((float4*)Al)[f] = ((const float4*)(K + base))[f];
  }
#endif
  for (int i = 0; i < 32; ++i) {
    int f = l + i * 32;
    float4 vv = ((const float4*)(V + base))[f];
    float4 sv;
    sv.x = (vv.x * 0.5f - 1.0f >= 0.0f) ? 1.0f : 0.0f;
    sv.y = (vv.y * 0.5f - 1.0f >= 0.0f) ? 1.0f : 0.0f;
    sv.z = (vv.z * 0.5f - 1.0f >= 0.0f) ? 1.0f : 0.0f;
    sv.w = (vv.w * 0.5f - 1.0f >= 0.0f) ? 1.0f : 0.0f;
    ((float4*)Vl)[f] = sv;
  }
#if USE_ASYNC_LDS
  wait_async0();
#endif
  __syncthreads();

  // KV = K^T * Vs
  for (int mw = 0; mw < 4; ++mw) {
    for (int nu = 0; nu < 4; ++nu) {
      v8f acc = {};
      #pragma unroll
      for (int kk = 0; kk < 16; ++kk) {
        int hb = kk * 4 + kb2;
        v2f a, b;
        a.x = Al[hb * 64 + mw * 16 + lm];         // A[m=w, k=h] = K[h][w]
        a.y = Al[(hb + 1) * 64 + mw * 16 + lm];
        b.x = Vl[hb * 64 + nu * 16 + lm];         // B[k=h, n=u] = Vs[h][u]
        b.y = Vl[(hb + 1) * 64 + nu * 16 + lm];
        acc = __builtin_amdgcn_wmma_f32_16x16x4_f32(
            false, a, false, b, (short)0, acc, false, false);
      }
      int w0 = mw * 16 + ro;
      #pragma unroll
      for (int r = 0; r < 8; ++r)
        KVl[(w0 + r) * 64 + nu * 16 + lm] = acc[r];
    }
  }
  __syncthreads();

  // reload Q over K's LDS (prior DS reads drained by the barrier above)
#if USE_ASYNC_LDS
  for (int i = 0; i < 32; ++i) {
    int f = l + i * 32;
    async_g2l_b128((const float4*)(Q + base) + f, (float4*)Al + f);
  }
  wait_async0();
#else
  for (int i = 0; i < 32; ++i)
    ((float4*)Al)[l + i * 32] = ((const float4*)(Q + base))[l + i * 32];
#endif
  __syncthreads();

  // A = Q * KV, scale 0.125, LIF(thr=0.5) -> binary map for proj RepConv
  for (int mh = 0; mh < 4; ++mh) {
    for (int nu = 0; nu < 4; ++nu) {
      v8f acc = {};
      #pragma unroll
      for (int kk = 0; kk < 16; ++kk) {
        int wb = kk * 4 + kb2;
        v2f a, b;
        a.x = Al[(mh * 16 + lm) * 64 + wb];       // A[m=h, k=w] = Q[h][w]
        a.y = Al[(mh * 16 + lm) * 64 + wb + 1];
        b.x = KVl[wb * 64 + nu * 16 + lm];        // B[k=w, n=u] = KV[w][u]
        b.y = KVl[(wb + 1) * 64 + nu * 16 + lm];
        acc = __builtin_amdgcn_wmma_f32_16x16x4_f32(
            false, a, false, b, (short)0, acc, false, false);
      }
      int h0 = mh * 16 + ro;
      #pragma unroll
      for (int r = 0; r < 8; ++r) {
        float v = acc[r] * 0.125f;
        // lif(a, 0.5): spike(a/2 - 0.5) => a >= 1.0
        out[base + (size_t)(h0 + r) * 64 + nu * 16 + lm] = (v >= 1.0f) ? 1.0f : 0.0f;
      }
    }
  }
}

// ---------------------------------------------------------------------------
// Host orchestration
// ---------------------------------------------------------------------------
namespace {

struct BNp { const float *g, *b, *m, *v; };
struct Rep { const float* w1; BNp bn1; const float* dw; const float* w2; BNp bn2; };
struct Blk {
  Rep q; BNp q_bn; Rep k; BNp k_bn; Rep v; BNp v_bn; Rep proj; BNp proj_bn;
  const float *mw1, *mb1; BNp mbn1; const float *mw2, *mb2; BNp mbn2;
};
struct ST { float* s; float* t; };

static void gemm(hipStream_t st, const float* in, int Ctot, int c_off, int C,
                 const unsigned short* W, const float* s, const float* t,
                 float* out, int Otot, int o_off, int O,
                 const float* res, int spike, float thr, int silu)
{
  dim3 grid(HW / 128, O / 128, BATCH);
  gemm_bf16_kernel<<<grid, 256, 0, st>>>(in, Ctot, c_off, C, W, s, t,
                                         out, Otot, o_off, res, spike, thr, silu);
}

} // namespace

extern "C" void kernel_launch(void* const* d_in, const int* in_sizes, int n_in,
                              void* d_out, int out_size, void* d_ws, size_t ws_size,
                              hipStream_t stream)
{
  (void)in_sizes; (void)n_in; (void)out_size; (void)ws_size;

  // ---- walk flattened params in setup_inputs() dict-insertion order ----
  int idx = 0;
  auto F = [&](void) -> const float* { return (const float*)d_in[idx++]; };
  auto rdBN = [&](void) -> BNp { BNp r; r.g = F(); r.b = F(); r.m = F(); r.v = F(); return r; };
  auto rdRep = [&](void) -> Rep {
    Rep r; r.w1 = F(); r.bn1 = rdBN(); r.dw = F(); r.w2 = F(); r.bn2 = rdBN(); return r;
  };

  const float* x     = F();
  const float* cv1_w = F();
  BNp cv1_bn = rdBN();
  Blk blk[4];
  for (int i = 0; i < 4; ++i) {
    Blk& b = blk[i];
    b.q = rdRep(); b.q_bn = rdBN();
    b.k = rdRep(); b.k_bn = rdBN();
    b.v = rdRep(); b.v_bn = rdBN();
    b.proj = rdRep(); b.proj_bn = rdBN();
    b.mw1 = F(); b.mb1 = F(); b.mbn1 = rdBN();
    b.mw2 = F(); b.mb2 = F(); b.mbn2 = rdBN();
  }
  const float* cv2_w = F();
  BNp cv2_bn = rdBN();

  // ---- workspace bump allocator (activations ~224 MB + params ~8 MB) ----
  char*  ws  = (char*)d_ws;
  size_t off = 0;
  auto alloc = [&](size_t bytes) -> void* {
    void* p = ws + off;
    off += (bytes + 255) & ~(size_t)255;
    return p;
  };
  const size_t PLANE = (size_t)BATCH * 256 * HW * 4;       // 33.5 MB
  float* Y  = (float*)alloc((size_t)BATCH * 512 * HW * 4); // persistent concat buf
  float* TA = (float*)alloc(PLANE);
  float* TB = (float*)alloc(PLANE);
  float* TQ = (float*)alloc(PLANE);
  float* TK = (float*)alloc(PLANE);
  float* TV = (float*)alloc(PLANE);

  auto cvtW = [&](const float* w, int n) -> const unsigned short* {
    unsigned short* d = (unsigned short*)alloc((size_t)n * 2);
    f32_to_bf16_kernel<<<(n + 255) / 256, 256, 0, stream>>>(w, d, n);
    return d;
  };
  auto fuse1 = [&](BNp bn, const float* bias, int C) -> ST {
    ST r; r.s = (float*)alloc(C * 4); r.t = (float*)alloc(C * 4);
    bn_fuse_kernel<<<(C + 255) / 256, 256, 0, stream>>>(
        bn.g, bn.b, bn.m, bn.v, bias,
        nullptr, nullptr, nullptr, nullptr, r.s, r.t, C);
    return r;
  };
  auto fuse2 = [&](BNp inner, BNp outer, int C) -> ST {
    ST r; r.s = (float*)alloc(C * 4); r.t = (float*)alloc(C * 4);
    bn_fuse_kernel<<<(C + 255) / 256, 256, 0, stream>>>(
        inner.g, inner.b, inner.m, inner.v, nullptr,
        outer.g, outer.b, outer.m, outer.v, r.s, r.t, C);
    return r;
  };

  const int DWGRID = BATCH * 256 * HW / 256;  // 32768

  // ================= cv1: conv1x1 512->512 + BN + SiLU -> Y ================
  {
    const unsigned short* w16 = cvtW(cv1_w, 512 * 512);
    ST st = fuse1(cv1_bn, nullptr, 512);
    gemm(stream, x, 512, 0, 512, w16, st.s, st.t, Y, 512, 0, 512,
         nullptr, 0, 0.0f, 1);
  }

  // ================= 4 spiking blocks on b-half (Y channels 256..511) ======
  for (int i = 0; i < 4; ++i) {
    Blk& b = blk[i];

    // --- ms_attn: q/k/v RepConv chains (input = spike(x/2 - 1)) ---
    const Rep* reps[3]  = { &b.q, &b.k, &b.v };
    BNp  outer[3]       = { b.q_bn, b.k_bn, b.v_bn };
    float* outs[3]      = { TQ, TK, TV };
    for (int j = 0; j < 3; ++j) {
      const Rep& r = *reps[j];
      const unsigned short* w1b = cvtW(r.w1, 256 * 256);
      ST s1 = fuse1(r.bn1, nullptr, 256);
      gemm(stream, Y, 512, 256, 256, w1b, s1.s, s1.t, TA, 256, 0, 256,
           nullptr, 1, 1.0f, 0);                               // spike prologue
      dwconv3x3_kernel<<<DWGRID, 256, 0, stream>>>(TA, r.dw, TB, 256);
      const unsigned short* w2b = cvtW(r.w2, 256 * 256);
      ST s2 = fuse2(r.bn2, outer[j], 256);                     // bn2 then q/k/v_bn
      gemm(stream, TB, 256, 0, 256, w2b, s2.s, s2.t, outs[j], 256, 0, 256,
           nullptr, 0, 0.0f, 0);
    }

    // --- spatial attention (f32 WMMA), spiked output in TA (binary) ---
    attn_kernel<<<BATCH * 256, 32, 0, stream>>>(TQ, TK, TV, TA);

    // --- proj RepConv on binary attention map, residual into Y@256 ---
    {
      const unsigned short* w1b = cvtW(b.proj.w1, 256 * 256);
      ST s1 = fuse1(b.proj.bn1, nullptr, 256);
      gemm(stream, TA, 256, 0, 256, w1b, s1.s, s1.t, TB, 256, 0, 256,
           nullptr, 0, 0.0f, 0);                               // input already binary
      dwconv3x3_kernel<<<DWGRID, 256, 0, stream>>>(TB, b.proj.dw, TK, 256);
      const unsigned short* w2b = cvtW(b.proj.w2, 256 * 256);
      ST s2 = fuse2(b.proj.bn2, b.proj_bn, 256);
      gemm(stream, TK, 256, 0, 256, w2b, s2.s, s2.t, Y, 512, 256, 256,
           Y, 0, 0.0f, 0);                                     // += x residual
    }

    // --- ms_mlp: two spiking conv1x1 with bias+BN, residual into Y@256 ---
    {
      const unsigned short* w1b = cvtW(b.mw1, 256 * 256);
      ST m1 = fuse1(b.mbn1, b.mb1, 256);                       // bias folded
      gemm(stream, Y, 512, 256, 256, w1b, m1.s, m1.t, TA, 256, 0, 256,
           nullptr, 1, 1.0f, 0);
      const unsigned short* w2b = cvtW(b.mw2, 256 * 256);
      ST m2 = fuse1(b.mbn2, b.mb2, 256);
      gemm(stream, TA, 256, 0, 256, w2b, m2.s, m2.t, Y, 512, 256, 256,
           Y, 1, 1.0f, 0);
    }
  }

  // ================= cv2: conv1x1 on concat(Y) + BN + SiLU -> d_out ========
  {
    const unsigned short* w16 = cvtW(cv2_w, 512 * 512);
    ST st = fuse1(cv2_bn, nullptr, 512);
    gemm(stream, Y, 512, 0, 512, w16, st.s, st.t, (float*)d_out, 512, 0, 512,
         nullptr, 0, 0.0f, 1);
  }
}
